// BinaryLinear_12549894438939
// MI455X (gfx1250) — matrix operand order repaired and checked by reading, not compiled, not run on hardware
//
#include <hip/hip_runtime.h>
#include <hip/hip_bf16.h>

// ---------------------------------------------------------------------------
// BinaryLinear forward:  out = x @ sign(W)^T
//   x: [32768, 2048] fp32, W: [2048, 2048] fp32, out: [32768, 2048] fp32
//
// Pass 1 (prep): binarize W -> bf16 {-1,0,+1} into workspace (8 MB, exact).
// Pass 2 (GEMM): split x into bf16 hi/lo (x = hi + lo; products vs +/-1 are
// exact), accumulate both passes with v_wmma_f32_16x16x32_bf16 in fp32.
// B tiles are copied global->LDS with CDNA5 async copies (ASYNCcnt).
// ---------------------------------------------------------------------------

typedef __attribute__((ext_vector_type(16))) __bf16 v16bf;
typedef __attribute__((ext_vector_type(8)))  float  v8f;

#define TOKENS 32768
#define KDIM   2048
#define NDIM   2048

#define BM 128
#define BN 128
#define BK 32
#define LDSW 40   // padded LDS row stride in bf16 elements (32 + 8 pad), 80 B

union FragU { v16bf v; uint4 q[2]; };
union H4    { __bf16 h[4]; uint2 u; };

__device__ __forceinline__ __bf16 bsign(float w) {
    unsigned u = __builtin_bit_cast(unsigned, w);
    unsigned short s = (u & 0x7fffffffu)
        ? (unsigned short)(0x3f80u | ((u >> 16) & 0x8000u))   // +/-1.0 bf16
        : (unsigned short)0;                                  // sign(0) = 0
    return __builtin_bit_cast(__bf16, s);
}

// gfx1250 async global->LDS copy (VGLOBAL GLOBAL_LOAD_ASYNC_TO_LDS_B128,
// tracked with ASYNCcnt). Generic-pointer low 32 bits == LDS byte offset.
__device__ __forceinline__ void async_g2l_b128(const void* gptr, void* lptr) {
    unsigned l = (unsigned)(uintptr_t)lptr;
    asm volatile("global_load_async_to_lds_b128 %0, %1, off"
                 :: "v"(l), "v"(gptr) : "memory");
}
__device__ __forceinline__ void wait_async0() {
    asm volatile("s_wait_asynccnt 0x0" ::: "memory");
}

// ---------------------------------------------------------------------------
// Pass 1: W (fp32) -> sign(W) as bf16, done ONCE instead of per M-block.
// ---------------------------------------------------------------------------
__global__ __launch_bounds__(256) void prep_w_kernel(const float* __restrict__ W,
                                                     __bf16* __restrict__ Ws) {
    const int base = (blockIdx.x * 256 + threadIdx.x) * 16;
#pragma unroll
    for (int c = 0; c < 4; ++c) {
        float4 w4 = *reinterpret_cast<const float4*>(W + base + c * 4);
        H4 s;
        s.h[0] = bsign(w4.x); s.h[1] = bsign(w4.y);
        s.h[2] = bsign(w4.z); s.h[3] = bsign(w4.w);
        *reinterpret_cast<uint2*>(Ws + base + c * 4) = s.u;
    }
}

// ---------------------------------------------------------------------------
// Pass 2: tiled WMMA GEMM. USE_WS=true reads pre-binarized bf16 weights via
// async global->LDS copies; USE_WS=false is the fully fused fallback.
// ---------------------------------------------------------------------------
template <bool USE_WS>
__global__ __launch_bounds__(256) void binlin_wmma_kernel(
        const float*  __restrict__ X,
        const float*  __restrict__ W,
        const __bf16* __restrict__ Ws,
        float*        __restrict__ O) {
    __shared__ __bf16 sAhi[2][BM * LDSW];
    __shared__ __bf16 sAlo[2][BM * LDSW];
    __shared__ __bf16 sBsn[2][BM * LDSW];

    const int t      = threadIdx.x;
    const int lane   = t & 31;
    const int wave   = t >> 5;
    const int wm     = wave >> 2;        // 0..1 : 64-row strip of the tile
    const int wn     = wave & 3;         // 0..3 : 32-col strip of the tile
    const int laneLo = lane & 15;
    const int laneHi = lane >> 4;

    const int mBase = blockIdx.y * BM;
    const int nBase = blockIdx.x * BN;

    const float*  Ag  = X  + (size_t)mBase * KDIM;
    const float*  Bg  = W  + (size_t)nBase * KDIM;
    const __bf16* Bsg = Ws + (size_t)nBase * KDIM;

    // A staging coords: 4 float4 chunks (16 floats / thread / k-tile).
    int lrow[4], lkq[4];
#pragma unroll
    for (int i = 0; i < 4; ++i) {
        int idx = t + i * 256;           // 0..1023
        lrow[i] = idx >> 3;              // 0..127
        lkq[i]  = (idx & 7) * 4;         // K offset (floats)
    }
    // B async staging coords (USE_WS): 2 x b128 chunks (16 bf16 / thread).
    int brow[2], bkq[2];
#pragma unroll
    for (int i = 0; i < 2; ++i) {
        int idx = t + i * 256;           // 0..511
        brow[i] = idx >> 2;              // 0..127
        bkq[i]  = (idx & 3) * 8;         // K offset (halves), 16 B chunks
    }

    v8f acc[4][2];
#pragma unroll
    for (int f = 0; f < 4; ++f)
#pragma unroll
        for (int g = 0; g < 2; ++g)
            acc[f][g] = (v8f){0.f, 0.f, 0.f, 0.f, 0.f, 0.f, 0.f, 0.f};

    // ---- prologue: stage k-tile 0 into buffer 0 ----
    float4 aPre[4], bPre[4];
    if (USE_WS) {
#pragma unroll
        for (int i = 0; i < 2; ++i)
            async_g2l_b128(Bsg + (size_t)brow[i] * KDIM + bkq[i],
                           &sBsn[0][brow[i] * LDSW + bkq[i]]);
    } else {
#pragma unroll
        for (int i = 0; i < 4; ++i)
            bPre[i] = *reinterpret_cast<const float4*>(Bg + (size_t)lrow[i] * KDIM + lkq[i]);
    }
#pragma unroll
    for (int i = 0; i < 4; ++i)
        aPre[i] = *reinterpret_cast<const float4*>(Ag + (size_t)lrow[i] * KDIM + lkq[i]);
#pragma unroll
    for (int i = 0; i < 4; ++i) {
        float av[4] = {aPre[i].x, aPre[i].y, aPre[i].z, aPre[i].w};
        H4 hh, ll;
#pragma unroll
        for (int j = 0; j < 4; ++j) {
            __bf16 h = (__bf16)av[j];
            hh.h[j] = h;
            ll.h[j] = (__bf16)(av[j] - (float)h);
        }
        int o = lrow[i] * LDSW + lkq[i];
        *reinterpret_cast<uint2*>(&sAhi[0][o]) = hh.u;
        *reinterpret_cast<uint2*>(&sAlo[0][o]) = ll.u;
        if (!USE_WS) {
            float bv[4] = {bPre[i].x, bPre[i].y, bPre[i].z, bPre[i].w};
            H4 ss;
#pragma unroll
            for (int j = 0; j < 4; ++j) ss.h[j] = bsign(bv[j]);
            *reinterpret_cast<uint2*>(&sBsn[0][o]) = ss.u;
        }
    }
    if (USE_WS) wait_async0();
    __syncthreads();

    // Fragment LDS addressing (documented CDNA5 WMMA VGPR layouts):
    //   A 16x32: lane<16 -> K {0..7, 16..23}; lane>=16 -> K {8..15, 24..31}
    //   B 32x16: lane<16 -> K {0..7, 16..23}; lane>=16 -> K {8..15, 24..31}
    const int aHalf0 = laneHi * 8;
    const int bHalf0 = laneHi * 8;

    const int KT = KDIM / BK;            // 64 k-tiles
    for (int kt = 0; kt < KT; ++kt) {
        const int  cur = kt & 1;
        const int  nxt = cur ^ 1;
        const bool pf  = (kt + 1) < KT;

        // Prefetch next k-tile early: async B straight into LDS, A into regs.
        if (pf) {
            const int k0 = (kt + 1) * BK;
            if (USE_WS) {
#pragma unroll
                for (int i = 0; i < 2; ++i)
                    async_g2l_b128(Bsg + (size_t)brow[i] * KDIM + k0 + bkq[i],
                                   &sBsn[nxt][brow[i] * LDSW + bkq[i]]);
            } else {
#pragma unroll
                for (int i = 0; i < 4; ++i)
                    bPre[i] = *reinterpret_cast<const float4*>(Bg + (size_t)lrow[i] * KDIM + k0 + lkq[i]);
            }
#pragma unroll
            for (int i = 0; i < 4; ++i)
                aPre[i] = *reinterpret_cast<const float4*>(Ag + (size_t)lrow[i] * KDIM + k0 + lkq[i]);
        }

        // ---- compute on current buffer: 16 WMMAs per wave per k-tile ----
        FragU bfr[2];
#pragma unroll
        for (int g = 0; g < 2; ++g) {
            int r = (wn * 32 + g * 16 + laneLo) * LDSW + bHalf0;
            bfr[g].q[0] = *reinterpret_cast<const uint4*>(&sBsn[cur][r]);
            bfr[g].q[1] = *reinterpret_cast<const uint4*>(&sBsn[cur][r + 16]);
        }
#pragma unroll
        for (int f = 0; f < 4; ++f) {
            int r = (wm * 64 + f * 16 + laneLo) * LDSW + aHalf0;
            FragU ah, al;
            ah.q[0] = *reinterpret_cast<const uint4*>(&sAhi[cur][r]);
            ah.q[1] = *reinterpret_cast<const uint4*>(&sAhi[cur][r + 16]);
            al.q[0] = *reinterpret_cast<const uint4*>(&sAlo[cur][r]);
            al.q[1] = *reinterpret_cast<const uint4*>(&sAlo[cur][r + 16]);
#pragma unroll
            for (int g = 0; g < 2; ++g) {
                acc[f][g] = __builtin_amdgcn_wmma_f32_16x16x32_bf16(
                    false, al.v, false, bfr[g].v, (short)0, acc[f][g], false, false);
                acc[f][g] = __builtin_amdgcn_wmma_f32_16x16x32_bf16(
                    false, ah.v, false, bfr[g].v, (short)0, acc[f][g], false, false);
            }
        }

        // ---- convert prefetched A (and B in fallback) into next buffer ----
        if (pf) {
#pragma unroll
            for (int i = 0; i < 4; ++i) {
                float av[4] = {aPre[i].x, aPre[i].y, aPre[i].z, aPre[i].w};
                H4 hh, ll;
#pragma unroll
                for (int j = 0; j < 4; ++j) {
                    __bf16 h = (__bf16)av[j];
                    hh.h[j] = h;
                    ll.h[j] = (__bf16)(av[j] - (float)h);
                }
                int o = lrow[i] * LDSW + lkq[i];
                *reinterpret_cast<uint2*>(&sAhi[nxt][o]) = hh.u;
                *reinterpret_cast<uint2*>(&sAlo[nxt][o]) = ll.u;
                if (!USE_WS) {
                    float bv[4] = {bPre[i].x, bPre[i].y, bPre[i].z, bPre[i].w};
                    H4 ss;
#pragma unroll
                    for (int j = 0; j < 4; ++j) ss.h[j] = bsign(bv[j]);
                    *reinterpret_cast<uint2*>(&sBsn[nxt][o]) = ss.u;
                }
            }
            if (USE_WS) wait_async0();
        }
        __syncthreads();
    }

    // ---- epilogue: C/D 16x16 layout (M = r + 8*laneHi, N = laneLo) ----
    // Nontemporal: 256 MB of output is never re-read; keep it out of L2.
#pragma unroll
    for (int f = 0; f < 4; ++f) {
#pragma unroll
        for (int g = 0; g < 2; ++g) {
            const int n  = nBase + wn * 32 + g * 16 + laneLo;
            const int mB = mBase + wm * 64 + f * 16 + laneHi * 8;
#pragma unroll
            for (int r = 0; r < 8; ++r) {
                __builtin_nontemporal_store(acc[f][g][r],
                                            &O[(size_t)(mB + r) * NDIM + n]);
            }
        }
    }
}

extern "C" void kernel_launch(void* const* d_in, const int* in_sizes, int n_in,
                              void* d_out, int out_size, void* d_ws, size_t ws_size,
                              hipStream_t stream) {
    const float* x = (const float*)d_in[0];   // [32768, 2048]
    const float* w = (const float*)d_in[1];   // [2048, 2048]
    float* out = (float*)d_out;               // [32768, 2048]

    dim3 grid(NDIM / BN, TOKENS / BM);        // (16, 256)
    const size_t ws_need = (size_t)KDIM * NDIM * sizeof(__bf16);   // 8 MB

    if (ws_size >= ws_need) {
        __bf16* Ws = (__bf16*)d_ws;
        prep_w_kernel<<<(KDIM * NDIM) / (256 * 16), 256, 0, stream>>>(w, Ws);
        binlin_wmma_kernel<true><<<grid, 256, 0, stream>>>(x, w, Ws, out);
    } else {
        binlin_wmma_kernel<false><<<grid, 256, 0, stream>>>(x, w, nullptr, out);
    }
}
